// CausalSelfAttention_67147518705858
// MI455X (gfx1250) — matrix operand order repaired and checked by reading, not compiled, not run on hardware
//
#include <hip/hip_runtime.h>
#include <hip/hip_bf16.h>

// ---------------------------------------------------------------------------
// Causal self-attention, MI455X (gfx1250), wave32 + WMMA bf16 (f32 accumulate)
//   1) qkv  = x @ W_qkv^T              (bf16 out, f32 acc, double-buffered)
//   2) attn = flash-attention(q,k,v)   (TDM K staging, permlane16 softmax)
//   3) out  = attn @ W_proj^T          (f32 out)
// ---------------------------------------------------------------------------

typedef __attribute__((ext_vector_type(16))) __bf16 v16bf;
typedef __attribute__((ext_vector_type(8)))  __bf16 v8bf;
typedef __attribute__((ext_vector_type(4)))  __bf16 v4bf;
typedef __attribute__((ext_vector_type(8)))  float  v8f;
typedef __attribute__((ext_vector_type(4)))  float  v4f;
typedef __attribute__((ext_vector_type(4)))  unsigned int u32x4;
typedef __attribute__((ext_vector_type(8)))  int i32x8;
typedef __attribute__((ext_vector_type(4)))  int i32x4;

#define D_MODEL 1024
#define N_HEADS 16
#define D_HEAD  64
#define SEQ_T   2048
#define BATCH   4

static __device__ __forceinline__ v8f wmma_bf16(v16bf a, v16bf b, v8f c) {
  return __builtin_amdgcn_wmma_f32_16x16x32_bf16(false, a, false, b, (short)0, c,
                                                 false, false);
}

// A fragment: 16(M) x 32(K), source row-major [16][ld] bf16.
static __device__ __forceinline__ v16bf load_fragA(const __bf16* base, int ld) {
  const int lane = threadIdx.x & 31;
  const int row = lane & 15;
  const int kb = (lane >> 4) << 3;
  const __bf16* p = base + row * ld + kb;
  v8bf lo = *(const v8bf*)(p);        // K = kb .. kb+7
  v8bf hi = *(const v8bf*)(p + 16);   // K = kb+16 .. kb+23
  return __builtin_shufflevector(lo, hi, 0, 1, 2, 3, 4, 5, 6, 7,
                                 8, 9, 10, 11, 12, 13, 14, 15);
}

// B fragment: 32(K) x 16(N), matrix supplied as [N][ld] row-major.
static __device__ __forceinline__ v16bf load_fragB(const __bf16* base, int ld) {
  const int lane = threadIdx.x & 31;
  const int n = lane & 15;
  const int kh = (lane >> 4) << 3;
  const __bf16* p = base + n * ld + kh;
  v8bf lo = *(const v8bf*)(p);
  v8bf hi = *(const v8bf*)(p + 16);
  return __builtin_shufflevector(lo, hi, 0, 1, 2, 3, 4, 5, 6, 7,
                                 8, 9, 10, 11, 12, 13, 14, 15);
}

// ---------------------------------------------------------------------------
// 16-lane xor shuffles via v_permlane16_b32 (pure VALU, no LDS traffic).
// ---------------------------------------------------------------------------
#if __has_builtin(__builtin_amdgcn_permlane16)
#define PERMLANE16(o, s, a, b) __builtin_amdgcn_permlane16((o), (s), (a), (b), false, false)
#define HAVE_PERMLANE 1
#elif __has_builtin(__builtin_amdgcn_permlane16_b32)
#define PERMLANE16(o, s, a, b) __builtin_amdgcn_permlane16_b32((o), (s), (a), (b), false, false)
#define HAVE_PERMLANE 1
#else
#define HAVE_PERMLANE 0
#endif

static constexpr unsigned xor_sel_lo(int m) {
  unsigned s = 0;
  for (int i = 0; i < 8; ++i) s |= ((unsigned)((i ^ m) & 15)) << (4 * i);
  return s;
}
static constexpr unsigned xor_sel_hi(int m) {
  unsigned s = 0;
  for (int i = 8; i < 16; ++i) s |= ((unsigned)((i ^ m) & 15)) << (4 * (i - 8));
  return s;
}

template <int M>
static __device__ __forceinline__ float xlane16(float x) {
#if HAVE_PERMLANE
  constexpr unsigned lo = xor_sel_lo(M), hi = xor_sel_hi(M);
  const int v = __float_as_int(x);
  return __int_as_float(PERMLANE16(v, v, (int)lo, (int)hi));
#else
  return __shfl_xor(x, M, 32);
#endif
}

static __device__ __forceinline__ float red_max16(float x) {
  x = fmaxf(x, xlane16<1>(x));
  x = fmaxf(x, xlane16<2>(x));
  x = fmaxf(x, xlane16<4>(x));
  x = fmaxf(x, xlane16<8>(x));
  return x;
}
static __device__ __forceinline__ float red_sum16(float x) {
  x += xlane16<1>(x);
  x += xlane16<2>(x);
  x += xlane16<4>(x);
  x += xlane16<8>(x);
  return x;
}

// ---------------------------------------------------------------------------
// TDM 2D bf16 tile load (6-arg builtin on this toolchain).
// ---------------------------------------------------------------------------
#if __has_builtin(__builtin_amdgcn_tensor_load_to_lds)
#define HAVE_TDM 1
static __device__ __forceinline__ void tdm_load_2d_bf16(
    unsigned int lds_off, const __bf16* gaddr, unsigned int tile_d0,
    unsigned int tile_d1, unsigned int tensor_d0, unsigned int tensor_d1,
    unsigned long long stride0) {
  unsigned long long ga = (unsigned long long)(uintptr_t)gaddr;
  u32x4 g0;
  g0[0] = 1u;                                   // count=1 (user descriptor)
  g0[1] = lds_off;                              // lds_addr [63:32]
  g0[2] = (unsigned int)(ga & 0xFFFFFFFFu);     // global_addr[31:0]  -> [95:64]
  g0[3] = (unsigned int)((ga >> 32) & 0x1FFFFFFu) | (2u << 30);  // [120:96]+type=2
  i32x8 g1;
  g1[0] = (int)(1u << 16);                      // wg_mask=0, data_size=1 (2B)
  g1[1] = (int)((tensor_d0 & 0xFFFFu) << 16);   // tensor_dim0[15:0]
  g1[2] = (int)(((tensor_d0 >> 16) & 0xFFFFu) | ((tensor_d1 & 0xFFFFu) << 16));
  g1[3] = (int)(((tensor_d1 >> 16) & 0xFFFFu) | ((tile_d0 & 0xFFFFu) << 16));
  g1[4] = (int)(tile_d1 & 0xFFFFu);             // tile_dim1; tile_dim2=0
  g1[5] = (int)(unsigned int)(stride0 & 0xFFFFFFFFu);       // dim0_stride lo
  g1[6] = (int)(unsigned int)((stride0 >> 32) & 0xFFFFu);   // dim0_stride hi
  g1[7] = 0;
  i32x4 z4 = {0, 0, 0, 0};
  i32x8 z8 = {0, 0, 0, 0, 0, 0, 0, 0};
  __builtin_amdgcn_tensor_load_to_lds(g0, g1, z4, z4, z8, 0);
}
static __device__ __forceinline__ void tdm_wait() {
#if __has_builtin(__builtin_amdgcn_s_wait_tensorcnt)
  __builtin_amdgcn_s_wait_tensorcnt(0);
#else
  asm volatile("s_wait_tensorcnt 0" ::: "memory");
#endif
}
#else
#define HAVE_TDM 0
#endif

// ---------------------------------------------------------------------------
// GEMM: C[m][n] = sum_k A[m][k] * W[n][k]  (A @ W^T)
// Block tile 256(M) x 128(N), K-step 32, 8 waves in 4x2 grid, each wave a
// 64x64 tile (4x4 WMMA, 16 wmma : 16 ds_load per K-step -- frags reused 4x).
// Double-buffered LDS with register staging: global loads for tile t+1
// overlap WMMA on tile t.
// ---------------------------------------------------------------------------
template <int A_BF16, int C_BF16>
__global__ __launch_bounds__(256) void gemm_xwT_kernel(
    const void* __restrict__ Ap, const float* __restrict__ Bw,
    void* __restrict__ Cp, int M, int N, int K) {
  __shared__ __align__(16) __bf16 As[2][256][32];  // 32 KB
  __shared__ __align__(16) __bf16 Bs[2][128][32];  // 16 KB

  const int tid = threadIdx.x;
  const int lane = tid & 31;
  const int wid = tid >> 5;
  const int wm = wid >> 1;  // 0..3 -> 64-row slice
  const int wn = wid & 1;   // 0..1 -> 64-col slice
  const int m0 = blockIdx.y * 256;
  const int n0 = blockIdx.x * 128;

  const v8f vzero = {0.f, 0.f, 0.f, 0.f, 0.f, 0.f, 0.f, 0.f};
  v8f acc[4][4];
#pragma unroll
  for (int i = 0; i < 4; ++i)
#pragma unroll
    for (int j = 0; j < 4; ++j) acc[i][j] = vzero;

  // staging: A row = tid (all 32 cols); B row = tid>>1, half = (tid&1)*16
  const int br = tid >> 1;
  const int bh = (tid & 1) * 16;

  v8bf a8[4];
  v4f a4[8], b4[4];

  auto fetch = [&](int k0) {
    if (A_BF16) {
      const __bf16* ga = (const __bf16*)Ap + (size_t)(m0 + tid) * K + k0;
#pragma unroll
      for (int i = 0; i < 4; ++i) a8[i] = *(const v8bf*)(ga + 8 * i);
    } else {
      const float* ga = (const float*)Ap + (size_t)(m0 + tid) * K + k0;
#pragma unroll
      for (int i = 0; i < 8; ++i) a4[i] = *(const v4f*)(ga + 4 * i);
    }
    const float* gb = Bw + (size_t)(n0 + br) * K + k0 + bh;
#pragma unroll
    for (int i = 0; i < 4; ++i) b4[i] = *(const v4f*)(gb + 4 * i);
  };

  auto stash = [&](int buf) {
    if (A_BF16) {
#pragma unroll
      for (int i = 0; i < 4; ++i) *(v8bf*)&As[buf][tid][8 * i] = a8[i];
    } else {
#pragma unroll
      for (int i = 0; i < 8; ++i)
        *(v4bf*)&As[buf][tid][4 * i] = __builtin_convertvector(a4[i], v4bf);
    }
#pragma unroll
    for (int i = 0; i < 4; ++i)
      *(v4bf*)&Bs[buf][br][bh + 4 * i] = __builtin_convertvector(b4[i], v4bf);
  };

  const int nk = K >> 5;
  fetch(0);
  stash(0);
  __syncthreads();

  for (int t = 0; t < nk; ++t) {
    const int buf = t & 1;
    if (t + 1 < nk) fetch((t + 1) << 5);  // global loads in flight over compute

    v16bf af[4], bfg[4];
#pragma unroll
    for (int tm = 0; tm < 4; ++tm)
      af[tm] = load_fragA(&As[buf][wm * 64 + tm * 16][0], 32);
#pragma unroll
    for (int tn = 0; tn < 4; ++tn)
      bfg[tn] = load_fragB(&Bs[buf][wn * 64 + tn * 16][0], 32);
#pragma unroll
    for (int tn = 0; tn < 4; ++tn)
#pragma unroll
      for (int tm = 0; tm < 4; ++tm)
        acc[tm][tn] = wmma_bf16(af[tm], bfg[tn], acc[tm][tn]);

    if (t + 1 < nk) stash(1 - buf);
    __syncthreads();
  }

  const int col = lane & 15;
  const int rg = (lane >> 4) << 3;
#pragma unroll
  for (int tm = 0; tm < 4; ++tm)
#pragma unroll
    for (int tn = 0; tn < 4; ++tn)
#pragma unroll
      for (int e = 0; e < 8; ++e) {
        const int m = m0 + wm * 64 + tm * 16 + rg + e;
        const int n = n0 + wn * 64 + tn * 16 + col;
        const float v = acc[tm][tn][e];
        if (C_BF16)
          ((__bf16*)Cp)[(size_t)m * N + n] = (__bf16)v;
        else
          ((float*)Cp)[(size_t)m * N + n] = v;
      }
}

// ---------------------------------------------------------------------------
// Flash attention: grid = (T/128, B*H), block = 256 (8 waves).
// K tile staged by the Tensor Data Mover (wave 0); V staged transposed by VALU.
// Q is pre-scaled by 1/sqrt(Dh)=0.125 (exact in bf16).
// ---------------------------------------------------------------------------
__global__ __launch_bounds__(256) void flash_attn_kernel(
    const __bf16* __restrict__ qkv, __bf16* __restrict__ outb) {
  __shared__ __align__(16) __bf16 Kt[32][64];      // [key][dh]
  __shared__ __align__(16) __bf16 Vt[64][32];      // transposed: [dh][key]
  __shared__ __align__(16) __bf16 Pt[8][16][32];   // per-wave P tile

  const int tid = threadIdx.x;
  const int lane = tid & 31;
  const int wid = tid >> 5;
  const int b = blockIdx.y >> 4;
  const int h = blockIdx.y & 15;
  const int qbase = blockIdx.x * 128 + wid * 16;

  const __bf16* qptr =
      qkv + (size_t)(b * SEQ_T + qbase) * (3 * D_MODEL) + h * D_HEAD;
  v16bf qf0 = load_fragA(qptr, 3 * D_MODEL);
  v16bf qf1 = load_fragA(qptr + 32, 3 * D_MODEL);
#pragma unroll
  for (int i = 0; i < 16; ++i) {  // fold softmax scale into Q (exact: 2^-3)
    qf0[i] = (__bf16)((float)qf0[i] * 0.125f);
    qf1[i] = (__bf16)((float)qf1[i] * 0.125f);
  }

  const v8f vzero = {0.f, 0.f, 0.f, 0.f, 0.f, 0.f, 0.f, 0.f};
  float mrow[8], lrow[8];
  v8f O[4];
#pragma unroll
  for (int e = 0; e < 8; ++e) { mrow[e] = -1e30f; lrow[e] = 0.f; }
#pragma unroll
  for (int t = 0; t < 4; ++t) O[t] = vzero;

  const int nch = blockIdx.x * 4 + 4;
  const int sr = tid >> 3;       // staging key row 0..31
  const int sc = (tid & 7) * 8;  // staging dh col base
  for (int c = 0; c < nch; ++c) {
    const int kc = c * 32;
    __syncthreads();
    {
      const size_t rowoff = (size_t)(b * SEQ_T + kc + sr) * (3 * D_MODEL);
#if HAVE_TDM
      if (wid == 0) {
        const __bf16* kbase =
            qkv + (size_t)(b * SEQ_T + kc) * (3 * D_MODEL) + D_MODEL + h * D_HEAD;
        tdm_load_2d_bf16((unsigned int)(uintptr_t)&Kt[0][0], kbase,
                         /*tile_d0=*/64, /*tile_d1=*/32,
                         /*tensor_d0=*/64, /*tensor_d1=*/SEQ_T,
                         /*stride0=*/3 * D_MODEL);
      }
#else
      const __bf16* kp = qkv + rowoff + D_MODEL + h * D_HEAD + sc;
      *(v8bf*)&Kt[sr][sc] = *(const v8bf*)kp;
#endif
      const __bf16* vp = qkv + rowoff + 2 * D_MODEL + h * D_HEAD + sc;
      v8bf vv = *(const v8bf*)vp;
#pragma unroll
      for (int j = 0; j < 8; ++j) Vt[sc + j][sr] = vv[j];  // transpose to LDS
#if HAVE_TDM
      if (wid == 0) tdm_wait();  // TENSORcnt==0 before block barrier
#endif
    }
    __syncthreads();
    if (kc > qbase + 15) continue;  // fully masked chunk; barriers stay uniform

    // S = Q @ K^T  -- hoist fragment loads, then 4 WMMAs
    v16bf kb0 = load_fragB(&Kt[0][0], 64);
    v16bf kb1 = load_fragB(&Kt[0][32], 64);
    v16bf kb2 = load_fragB(&Kt[16][0], 64);
    v16bf kb3 = load_fragB(&Kt[16][32], 64);
    v8f S0 = vzero, S1 = vzero;
    S0 = wmma_bf16(qf0, kb0, S0);
    S0 = wmma_bf16(qf1, kb1, S0);
    S1 = wmma_bf16(qf0, kb2, S1);
    S1 = wmma_bf16(qf1, kb3, S1);

    // V fragments issued now so DS latency hides under softmax VALU work
    v16bf vf0 = load_fragB(&Vt[0][0], 32);
    v16bf vf1 = load_fragB(&Vt[16][0], 32);
    v16bf vf2 = load_fragB(&Vt[32][0], 32);
    v16bf vf3 = load_fragB(&Vt[48][0], 32);

    const int col = lane & 15;
    const int rg = (lane >> 4) << 3;
#pragma unroll
    for (int e = 0; e < 8; ++e) {
      const int q = qbase + rg + e;
      const bool ok0 = (kc + col) <= q;
      const bool ok1 = (kc + 16 + col) <= q;
      const float s0 = S0[e];
      const float s1 = S1[e];
      float rmax = red_max16(fmaxf(ok0 ? s0 : -1e30f, ok1 ? s1 : -1e30f));
      const float mn = fmaxf(mrow[e], rmax);
      const float corr = __expf(mrow[e] - mn);
      const float p0 = ok0 ? __expf(s0 - mn) : 0.f;
      const float p1 = ok1 ? __expf(s1 - mn) : 0.f;
      const float rs = red_sum16(p0 + p1);
      lrow[e] = lrow[e] * corr + rs;
      mrow[e] = mn;
#pragma unroll
      for (int t = 0; t < 4; ++t) O[t][e] *= corr;
      Pt[wid][rg + e][col] = (__bf16)p0;  // re-layout C -> A via LDS
      Pt[wid][rg + e][col + 16] = (__bf16)p1;
    }
    asm volatile("s_wait_dscnt 0" ::: "memory");  // drain P stores (same wave)

    const v16bf pf = load_fragA(&Pt[wid][0][0], 32);
    O[0] = wmma_bf16(pf, vf0, O[0]);
    O[1] = wmma_bf16(pf, vf1, O[1]);
    O[2] = wmma_bf16(pf, vf2, O[2]);
    O[3] = wmma_bf16(pf, vf3, O[3]);
  }

  const int col = lane & 15;
  const int rg = (lane >> 4) << 3;
#pragma unroll
  for (int e = 0; e < 8; ++e) {
    const float inv = 1.f / lrow[e];
    __bf16* op =
        outb + (size_t)(b * SEQ_T + qbase + rg + e) * D_MODEL + h * D_HEAD;
#pragma unroll
    for (int t = 0; t < 4; ++t) op[t * 16 + col] = (__bf16)(O[t][e] * inv);
  }
}

// ---------------------------------------------------------------------------
extern "C" void kernel_launch(void* const* d_in, const int* in_sizes, int n_in,
                              void* d_out, int out_size, void* d_ws,
                              size_t ws_size, hipStream_t stream) {
  (void)in_sizes; (void)n_in; (void)out_size; (void)ws_size;
  const float* x = (const float*)d_in[0];      // [4,2048,1024]
  const float* Wqkv = (const float*)d_in[1];   // [3072,1024]
  const float* Wproj = (const float*)d_in[2];  // [1024,1024]

  __bf16* qkvb = (__bf16*)d_ws;                                // 8192*3072 bf16
  __bf16* attnb = qkvb + (size_t)BATCH * SEQ_T * 3 * D_MODEL;  // 8192*1024 bf16

  const int M = BATCH * SEQ_T;  // 8192

  gemm_xwT_kernel<0, 1><<<dim3(3 * D_MODEL / 128, M / 256), 256, 0, stream>>>(
      x, Wqkv, qkvb, M, 3 * D_MODEL, D_MODEL);

  flash_attn_kernel<<<dim3(SEQ_T / 128, BATCH * N_HEADS), 256, 0, stream>>>(
      qkvb, attnb);

  gemm_xwT_kernel<1, 0><<<dim3(D_MODEL / 128, M / 256), 256, 0, stream>>>(
      attnb, Wproj, d_out, M, D_MODEL, D_MODEL);
}
